// FeatureMultiHeadAttention_61426622267522
// MI455X (gfx1250) — matrix operand order repaired and checked by reading, not compiled, not run on hardware
//
#include <hip/hip_runtime.h>
#include <hip/hip_bf16.h>

// ---------------------------------------------------------------------------
// Fused multi-head attention for MI455X (gfx1250, wave32, WMMA bf16).
// One workgroup = (batch b, 32-row tile); loops all 8 heads so A_sum
// accumulates in LDS with no atomics and the concatenated head outputs feed
// the W-projection GEMM in the same kernel (zero global workspace).
//
// v3 changes:
//  - softmax stage re-mapped to 4-consecutive-column chunks per lane:
//    mask read as one uchar4 per chunk, S rows read/written as float4,
//    Asum accumulate vectorized (4x fewer ds/vmem ops in stage 2),
//  - Asum zero-init and final writeback as flat float4 copies.
// Carried over from v2: per-wave fixed mt with cached Q fragments, paired
// n-tiles sharing one A fragment per WMMA pair, padded LDS strides.
// LDS: S 66KB + Asum 64KB + Ocat 66KB = ~193KB (<320KB/WGP).
// ---------------------------------------------------------------------------

#define NH     8
#define HID    128
#define BSZ    32
#define SEQ    512
#define TM     32                  // rows of Q handled per workgroup
#define CATW   (NH * HID)          // 1024
#define SROW   (SEQ + 4)           // padded S row stride (floats)
#define CATS   (CATW + 8)          // padded Ocat row stride (bf16)
#define INV_SCALE 0.08838834764831845f   // 1 / sqrt(128)
#define NEG_BIG  (-1.0e30f)

typedef __bf16 bf16_t;
typedef __attribute__((ext_vector_type(16))) __bf16 bf16x16;
typedef __attribute__((ext_vector_type(8)))  float  f32x8;

// ---- fragment builders (CDNA5 16x16x32 bf16 WMMA register layouts) --------
// A matrix 16x32: lane l holds row m=l&15; VGPR v holds K pair at
//   k = (v>=4?16:0) + (l>=16?8:0) + 2*(v&3)
__device__ __forceinline__ bf16x16 frag_a_f32(const float* __restrict__ row,
                                              int kbase, int lane) {
  bf16x16 f;
  const int kh = (lane >> 4) << 3;
#pragma unroll
  for (int g = 0; g < 2; ++g)
#pragma unroll
    for (int r = 0; r < 4; ++r) {
      const int v  = g * 4 + r;
      const int kk = kbase + g * 16 + kh + r * 2;
      f[2 * v]     = (bf16_t)row[kk];
      f[2 * v + 1] = (bf16_t)row[kk + 1];
    }
  return f;
}

__device__ __forceinline__ bf16x16 frag_a_bf16(const bf16_t* __restrict__ row,
                                               int kbase, int lane) {
  bf16x16 f;
  const int kh = (lane >> 4) << 3;
#pragma unroll
  for (int g = 0; g < 2; ++g)
#pragma unroll
    for (int r = 0; r < 4; ++r) {
      const int v  = g * 4 + r;
      const int kk = kbase + g * 16 + kh + r * 2;
      f[2 * v]     = row[kk];
      f[2 * v + 1] = row[kk + 1];
    }
  return f;
}

// B matrix 32x16: lane l holds col n=l&15; VGPR v holds K pair at
//   k = (v>=4?16:0) + (l>=16?8:0) + 2*(v&3)
// Data for column n is a contiguous row (K^T and W^T GEMMs):
__device__ __forceinline__ bf16x16 frag_b_row_f32(const float* __restrict__ row,
                                                  int kbase, int lane) {
  bf16x16 f;
  const int kh = (lane >> 4) << 3;
#pragma unroll
  for (int v = 0; v < 8; ++v) {
    const int kk = kbase + ((v >> 2) << 4) + kh + (v & 3) * 2;
    f[2 * v]     = (bf16_t)row[kk];
    f[2 * v + 1] = (bf16_t)row[kk + 1];
  }
  return f;
}

// Column n strides through memory (V operand of the A@V GEMM):
__device__ __forceinline__ bf16x16 frag_b_col_f32(const float* __restrict__ col,
                                                  int kbase, int ld, int lane) {
  bf16x16 f;
  const int kh = (lane >> 4) << 3;
#pragma unroll
  for (int v = 0; v < 8; ++v) {
    const int kk = kbase + ((v >> 2) << 4) + kh + (v & 3) * 2;
    f[2 * v]     = (bf16_t)col[(size_t)kk * ld];
    f[2 * v + 1] = (bf16_t)col[(size_t)(kk + 1) * ld];
  }
  return f;
}

__device__ __forceinline__ f32x8 wmma_bf16(bf16x16 a, bf16x16 b, f32x8 c) {
  return __builtin_amdgcn_wmma_f32_16x16x32_bf16(
      /*neg_a=*/false, a, /*neg_b=*/false, b,
      /*c_mod=*/(short)0, c, /*reuse_a=*/false, /*reuse_b=*/false);
}

// ---------------------------------------------------------------------------
extern "C" __global__ void __launch_bounds__(256, 1)
fmha_fused_kernel(const float* __restrict__ Q, const float* __restrict__ K,
                  const float* __restrict__ V, const float* __restrict__ W,
                  const float* __restrict__ Bias,
                  const unsigned char* __restrict__ Mask,
                  float* __restrict__ Out, float* __restrict__ AsumOut) {
  extern __shared__ float smem[];
  float*  S    = smem;                         // TM x SROW  (scores / probs)
  float*  Asum = smem + TM * SROW;             // TM x SEQ   (sum over heads)
  bf16_t* Ocat = (bf16_t*)(Asum + TM * SEQ);   // TM x CATS  (concat head outs)

  const int tid  = threadIdx.x;
  const int lane = tid & 31;
  const int wave = tid >> 5;
  const int b    = blockIdx.x;
  const int row0 = blockIdx.y * TM;
  const int lrow = lane & 15;          // row/col within a 16x16 tile
  const int lhi8 = (lane >> 4) << 3;   // C/D fragment row offset
  const int mt   = wave & 1;           // this wave's fixed M-tile (0..1)
  const int ntb  = wave >> 1;          // this wave's N-tile base (0..3)

  {
    const float4 z4 = {0.0f, 0.0f, 0.0f, 0.0f};
    float4* az = (float4*)Asum;
    for (int i = tid; i < TM * SEQ / 4; i += 256) az[i] = z4;
  }
  __syncthreads();

  for (int h = 0; h < NH; ++h) {
    const int bh = h * BSZ + b;
    const float* qh = Q + (size_t)bh * SEQ * HID;
    const float* kh = K + (size_t)bh * SEQ * HID;
    const float* vh = V + (size_t)bh * SEQ * HID;
    const unsigned char* mh = Mask + (size_t)bh * SEQ * SEQ;

    // ---- Stage 1: S = Q * K^T  (raw scores; scale+mask applied later) -----
    // Cache this wave's 4 Q fragments once, reuse for all 8 n-tiles.
    bf16x16 aq[4];
    {
      const float* qrow = qh + (size_t)(row0 + mt * 16 + lrow) * HID;
#pragma unroll
      for (int ks = 0; ks < 4; ++ks) aq[ks] = frag_a_f32(qrow, ks * 32, lane);
    }
#pragma unroll 1
    for (int j = 0; j < 8; ++j) {
      const int nt = ntb + j * 4;       // 0..31 over the 4 wave pairs
      const float* krow = kh + (size_t)(nt * 16 + lrow) * HID;
      f32x8 acc = {};
#pragma unroll
      for (int ks = 0; ks < 4; ++ks)
        acc = wmma_bf16(aq[ks], frag_b_row_f32(krow, ks * 32, lane), acc);
      const int n = nt * 16 + lrow;
#pragma unroll
      for (int vv = 0; vv < 8; ++vv)
        S[(mt * 16 + lhi8 + vv) * SROW + n] = acc[vv];
    }
    __syncthreads();

    // ---- Stage 2: scale + mask + row softmax + Asum (wave per row) --------
    // Lane owns 4 chunks of 4 consecutive columns: n = blk*128 + lane*4 + q.
#pragma unroll 1
    for (int rr = 0; rr < 4; ++rr) {
      const int m = wave + rr * 8;
      float* srow = S + m * SROW;
      const unsigned char* mrow = mh + (size_t)(row0 + m) * SEQ;
      float4 sv[4];
      float vmax = NEG_BIG;
#pragma unroll
      for (int blk = 0; blk < 4; ++blk) {
        const int n = blk * 128 + lane * 4;
        float4 s4 = *(const float4*)(srow + n);
        const uchar4 mk = *(const uchar4*)(mrow + n);
        s4.x = mk.x ? NEG_BIG : s4.x * INV_SCALE;
        s4.y = mk.y ? NEG_BIG : s4.y * INV_SCALE;
        s4.z = mk.z ? NEG_BIG : s4.z * INV_SCALE;
        s4.w = mk.w ? NEG_BIG : s4.w * INV_SCALE;
        sv[blk] = s4;
        vmax = fmaxf(vmax, fmaxf(fmaxf(s4.x, s4.y), fmaxf(s4.z, s4.w)));
      }
#pragma unroll
      for (int off = 16; off > 0; off >>= 1)
        vmax = fmaxf(vmax, __shfl_xor(vmax, off, 32));
      float vsum = 0.0f;
#pragma unroll
      for (int blk = 0; blk < 4; ++blk) {
        sv[blk].x = __expf(sv[blk].x - vmax);
        sv[blk].y = __expf(sv[blk].y - vmax);
        sv[blk].z = __expf(sv[blk].z - vmax);
        sv[blk].w = __expf(sv[blk].w - vmax);
        vsum += (sv[blk].x + sv[blk].y) + (sv[blk].z + sv[blk].w);
      }
#pragma unroll
      for (int off = 16; off > 0; off >>= 1) vsum += __shfl_xor(vsum, off, 32);
      const float inv = 1.0f / vsum;
      float* arow = Asum + m * SEQ;
#pragma unroll
      for (int blk = 0; blk < 4; ++blk) {
        const int n = blk * 128 + lane * 4;
        float4 a4;
        a4.x = sv[blk].x * inv;
        a4.y = sv[blk].y * inv;
        a4.z = sv[blk].z * inv;
        a4.w = sv[blk].w * inv;
        *(float4*)(srow + n) = a4;      // probs for stage 3
        float4 c4 = *(const float4*)(arow + n);
        c4.x += a4.x; c4.y += a4.y; c4.z += a4.z; c4.w += a4.w;
        *(float4*)(arow + n) = c4;      // same thread owns (m,n) every head
      }
    }
    __syncthreads();

    // ---- Stage 3: O_h = A * V ; one A fragment feeds two n-tiles ----------
    {
      const float* srow = S + (size_t)(mt * 16 + lrow) * SROW;
      const int d0 = ntb * 16 + lrow;         // 0..63
      const int d1 = d0 + 64;                 // 64..127
      const float* vc0 = vh + d0;
      const float* vc1 = vh + d1;
      f32x8 acc0 = {}, acc1 = {};
#pragma unroll 1
      for (int ks = 0; ks < 16; ++ks) {
        const bf16x16 a = frag_a_f32(srow, ks * 32, lane);
        acc0 = wmma_bf16(a, frag_b_col_f32(vc0, ks * 32, HID, lane), acc0);
        acc1 = wmma_bf16(a, frag_b_col_f32(vc1, ks * 32, HID, lane), acc1);
      }
#pragma unroll
      for (int vv = 0; vv < 8; ++vv) {
        const int m = mt * 16 + lhi8 + vv;
        Ocat[(size_t)m * CATS + h * HID + d0] = (bf16_t)acc0[vv];
        Ocat[(size_t)m * CATS + h * HID + d1] = (bf16_t)acc1[vv];
      }
    }
    __syncthreads();
  }

  // ---- Stage 4: write A_sum tile (dst rows contiguous -> flat f4 copy) ----
  {
    const float4* src = (const float4*)Asum;
    float4* dst = (float4*)(AsumOut + ((size_t)b * SEQ + row0) * SEQ);
    for (int i = tid; i < TM * SEQ / 4; i += 256) dst[i] = src[i];
  }

  // ---- Stage 5: projection Out = Ocat(32x1024) * W^T + bias ---------------
  {
    const bf16_t* arow = Ocat + (size_t)(mt * 16 + lrow) * CATS;
    const int j0 = ntb * 16 + lrow;          // 0..63
    const int j1 = j0 + 64;                  // 64..127
    const float* w0 = W + (size_t)j0 * CATW;
    const float* w1 = W + (size_t)j1 * CATW;
    f32x8 acc0 = {}, acc1 = {};
#pragma unroll 1
    for (int ks = 0; ks < 32; ++ks) {
      const bf16x16 a = frag_a_bf16(arow, ks * 32, lane);
      acc0 = wmma_bf16(a, frag_b_row_f32(w0, ks * 32, lane), acc0);
      acc1 = wmma_bf16(a, frag_b_row_f32(w1, ks * 32, lane), acc1);
    }
    const float b0 = Bias[j0];
    const float b1 = Bias[j1];
#pragma unroll
    for (int vv = 0; vv < 8; ++vv) {
      const int m = mt * 16 + lhi8 + vv;
      float* orow = Out + ((size_t)b * SEQ + row0 + m) * HID;
      orow[j0] = acc0[vv] + b0;
      orow[j1] = acc1[vv] + b1;
    }
  }
}

// ---------------------------------------------------------------------------
extern "C" void kernel_launch(void* const* d_in, const int* in_sizes, int n_in,
                              void* d_out, int out_size, void* d_ws, size_t ws_size,
                              hipStream_t stream) {
  (void)in_sizes; (void)n_in; (void)out_size; (void)d_ws; (void)ws_size;
  const float* q    = (const float*)d_in[0];
  const float* k    = (const float*)d_in[1];
  const float* v    = (const float*)d_in[2];
  const float* W    = (const float*)d_in[3];
  const float* bias = (const float*)d_in[4];
  const unsigned char* mask = (const unsigned char*)d_in[5];   // jax bool -> byte

  float* out   = (float*)d_out;
  float* a_sum = out + (size_t)BSZ * SEQ * HID;  // outputs concatenated flat

  const size_t smem = (size_t)TM * SROW * sizeof(float)      // S (padded)
                    + (size_t)TM * SEQ  * sizeof(float)      // Asum
                    + (size_t)TM * CATS * sizeof(bf16_t);    // Ocat (padded)
  (void)hipFuncSetAttribute((const void*)fmha_fused_kernel,
                            hipFuncAttributeMaxDynamicSharedMemorySize,
                            (int)smem);
  dim3 grid(BSZ, SEQ / TM);   // 32 x 16 = 512 workgroups, 8 waves each
  fmha_fused_kernel<<<grid, dim3(256, 1, 1), smem, stream>>>(
      q, k, v, W, bias, mask, out, a_sum);
}
